// MHInrAttn_17738214933128
// MI455X (gfx1250) — matrix operand order repaired and checked by reading, not compiled, not run on hardware
//
#include <hip/hip_runtime.h>
#include <hip/hip_bf16.h>
#include <math.h>

// ---------------------------------------------------------------------------
// MHA with structural softmax bias — MI455X (gfx1250), wave32, bf16 WMMA.
// Bandwidth-bound on str_mat (2 x 256MB reads); all GEMMs ride
// v_wmma_f32_16x16x32_bf16 with double-buffered LDS staging so the WMMA
// pipe overlaps global traffic. Staging uses global_load_async_to_lds_b128
// (builtin confirmed present; b128 = int4-typed pointers per clang).
// ---------------------------------------------------------------------------

typedef __bf16 bf16;
typedef __attribute__((ext_vector_type(16))) __bf16 v16bf;
typedef __attribute__((ext_vector_type(8)))  float  v8f;

#define N_HEAD 16
#define SEQ    1024
#define FEAT   1024
#define DHEAD  64
#define BATCH  4
#define ROWS_TOTAL (BATCH * SEQ)   // 4096

// -------- optional CDNA5 async global->LDS staging path --------------------
#if defined(__has_builtin)
#if __has_builtin(__builtin_amdgcn_global_load_async_to_lds_b128) && \
    __has_builtin(__builtin_amdgcn_s_wait_asynccnt)
#define USE_ASYNC_LDS 1
#endif
#endif
#ifndef USE_ASYNC_LDS
#define USE_ASYNC_LDS 0
#endif

#if USE_ASYNC_LDS
typedef int async_b128 __attribute__((vector_size(16)));
typedef __attribute__((address_space(1))) async_b128* g_b128_ptr;   // global src
typedef __attribute__((address_space(3))) async_b128* l_b128_ptr;   // LDS dst
#define WAIT_STAGE() __builtin_amdgcn_s_wait_asynccnt(0)
#else
#define WAIT_STAGE() ((void)0)
#endif

static __device__ inline v8f wmma_bf16(v16bf a, v16bf b, v8f c) {
  // D = A(16x32 bf16) * B(32x16 bf16) + C(16x16 f32)
  return __builtin_amdgcn_wmma_f32_16x16x32_bf16(
      /*neg_a=*/false, a, /*neg_b=*/false, b,
      /*c_mod=*/(short)0, c, /*reuse_a=*/false, /*reuse_b=*/false);
}

union frag_u { float4 f[2]; v16bf b; };

// A fragment: 16x32 (MxK) tile, row-major, 'stride' elements/row.
// lanes 0-15: M=lane, K 0..7 (elems 0-7) and 16..23 (elems 8-15)
// lanes 16-31: M=lane-16, K 8..15 and 24..31
static __device__ inline v16bf load_a16x32(const bf16* __restrict__ p0, int stride) {
  const int lane = threadIdx.x & 31;
  const int m  = lane & 15;
  const int kb = (lane >> 4) << 3;     // 0 or 8
  const bf16* p = p0 + m * stride + kb;
  frag_u u;
  u.f[0] = *(const float4*)p;          // K kb..kb+7
  u.f[1] = *(const float4*)(p + 16);   // K kb+16..kb+23
  return u.b;
}

// B fragment from K-contiguous (transposed) memory: element (k,n) at
// p0[n*stride + k].  lanes 0-15: N=lane, K 0..7 and 16..23; lanes 16-31:
// N=lane-16, K 8..15 and 24..31.  Two 16B loads per lane.
static __device__ inline v16bf load_b32x16_T(const bf16* __restrict__ p0, int stride) {
  const int lane = threadIdx.x & 31;
  const int n  = lane & 15;
  const int ko = (lane >> 4) << 3;     // 0 or 8
  const bf16* p = p0 + n * stride + ko;
  frag_u u;
  u.f[0] = *(const float4*)p;
  u.f[1] = *(const float4*)(p + 16);
  return u.b;
}

// ---------------------------------------------------------------------------
// Tile staging: 128 rows x 32 bf16 from global (row stride FEAT) into LDS
// (row stride 40). 256 threads, 2 x 16B slots each, fully unrolled.
// ---------------------------------------------------------------------------
#define XS_STRIDE 40   // 32 + 8 bf16 pad (80B; 16B multiple; banks step 20)
#define TILE_ELT (128 * XS_STRIDE)

static __device__ inline void stage_tile_128x32(const bf16* __restrict__ g,
                                                bf16* __restrict__ l) {
  const int tid  = threadIdx.x;
  const int row0 = tid >> 2;                 // 0..63
  const int q    = (tid & 3) * 8;            // 0,8,16,24
  const bf16* s0 = g + (size_t)row0 * FEAT + q;
  const bf16* s1 = g + (size_t)(row0 + 64) * FEAT + q;
  bf16* d0 = l + row0 * XS_STRIDE + q;
  bf16* d1 = l + (row0 + 64) * XS_STRIDE + q;
#if USE_ASYNC_LDS
  __builtin_amdgcn_global_load_async_to_lds_b128(
      (g_b128_ptr)(void*)s0, (l_b128_ptr)(void*)d0, 0, 0);
  __builtin_amdgcn_global_load_async_to_lds_b128(
      (g_b128_ptr)(void*)s1, (l_b128_ptr)(void*)d1, 0, 0);
#else
  float4 v0 = *(const float4*)s0;            // both loads in flight first
  float4 v1 = *(const float4*)s1;
  *(float4*)d0 = v0;
  *(float4*)d1 = v1;
#endif
}

// ---------------------------------------------------------------------------
// K0a: plain f32 -> bf16 conversion (X)
// ---------------------------------------------------------------------------
__global__ __launch_bounds__(256) void cvt_f32_bf16(const float* __restrict__ src,
                                                    bf16* __restrict__ dst, int n) {
  for (int i = blockIdx.x * 256 + threadIdx.x; i < n; i += gridDim.x * 256)
    dst[i] = (bf16)src[i];
}

// ---------------------------------------------------------------------------
// K0b: f32 -> bf16 with transpose (weights): dst[n*1024 + k] = src[k*1024 + n]
// ---------------------------------------------------------------------------
__global__ __launch_bounds__(256) void cvt_transpose_f32_bf16(const float* __restrict__ src,
                                                              bf16* __restrict__ dst) {
  for (int i = blockIdx.x * 256 + threadIdx.x; i < FEAT * FEAT; i += gridDim.x * 256) {
    int k = i >> 10, n = i & 1023;
    dst[(size_t)n * FEAT + k] = (bf16)src[i];
  }
}

// ---------------------------------------------------------------------------
// GEMM mainloop (shared by K1/K4): 128x128 tile, 8 waves (2x4), K stepped by
// 32 with double-buffered LDS. Weights pre-transposed (Wt[n][k]).
// ---------------------------------------------------------------------------
#define GEMM_MAIN(XPTR, WTPTR, ACC)                                            \
  stage_tile_128x32((XPTR), Xs);                                               \
  stage_tile_128x32((WTPTR), WsT);                                             \
  WAIT_STAGE();                                                                \
  __syncthreads();                                                             \
  for (int it = 0; it < FEAT / 32; ++it) {                                     \
    const int cur = it & 1, nxt = cur ^ 1;                                     \
    if (it + 1 < FEAT / 32) {                                                  \
      stage_tile_128x32((XPTR) + (it + 1) * 32, Xs + nxt * TILE_ELT);          \
      stage_tile_128x32((WTPTR) + (it + 1) * 32, WsT + nxt * TILE_ELT);        \
    }                                                                          \
    const bf16* Xc = Xs + cur * TILE_ELT;                                      \
    const bf16* Wc = WsT + cur * TILE_ELT;                                     \
    v16bf afrag[4], bfrag[2];                                                  \
    _Pragma("unroll")                                                          \
    for (int ms = 0; ms < 4; ++ms)                                             \
      afrag[ms] = load_a16x32(Xc + (wm * 64 + ms * 16) * XS_STRIDE, XS_STRIDE);\
    _Pragma("unroll")                                                          \
    for (int ns = 0; ns < 2; ++ns)                                             \
      bfrag[ns] = load_b32x16_T(Wc + (wn * 32 + ns * 16) * XS_STRIDE, XS_STRIDE);\
    _Pragma("unroll")                                                          \
    for (int ms = 0; ms < 4; ++ms)                                             \
      _Pragma("unroll")                                                        \
      for (int ns = 0; ns < 2; ++ns)                                           \
        ACC[ms][ns] = wmma_bf16(afrag[ms], bfrag[ns], ACC[ms][ns]);            \
    WAIT_STAGE();                                                              \
    __syncthreads();                                                           \
  }

// ---------------------------------------------------------------------------
// K1: QKV projection GEMM. Y = X @ W + bias.
//  - Q,K scattered into per-head layout  q[b,h,si,di] (si-major, 64 wide)
//  - V   scattered into per-head TRANSPOSED layout Vt[bh][di][si]
// grid (32, 8, 3); z selects Q/K/V.
// ---------------------------------------------------------------------------
__global__ __launch_bounds__(256) void qkv_gemm_kernel(
    const bf16* __restrict__ Xb,
    const bf16* __restrict__ WqT, const bf16* __restrict__ WkT, const bf16* __restrict__ WvT,
    const float* __restrict__ bq, const float* __restrict__ bk, const float* __restrict__ bv,
    bf16* __restrict__ Qh, bf16* __restrict__ Kh, bf16* __restrict__ Vt)
{
  __shared__ bf16 Xs [2 * TILE_ELT];   // X tile, row-major [m][k], double buf
  __shared__ bf16 WsT[2 * TILE_ELT];   // W tile, N-major  [n][k], double buf

  const int z = blockIdx.z;
  const bf16*  Wt   = (z == 0) ? WqT : (z == 1) ? WkT : WvT;
  const float* bias = (z == 0) ? bq  : (z == 1) ? bk  : bv;

  const int tid = threadIdx.x;
  const int wave = tid >> 5, lane = tid & 31;
  const int wm = wave >> 2, wn = wave & 3;           // 2x4 wave grid
  const int m0 = blockIdx.x * 128;
  const int n0 = blockIdx.y * 128;

  v8f acc[4][2];
#pragma unroll
  for (int ms = 0; ms < 4; ++ms)
#pragma unroll
    for (int ns = 0; ns < 2; ++ns)
#pragma unroll
      for (int e = 0; e < 8; ++e) acc[ms][ns][e] = 0.0f;

  const bf16* Xg = Xb + (size_t)m0 * FEAT;
  const bf16* Wg = Wt + (size_t)n0 * FEAT;
  GEMM_MAIN(Xg, Wg, acc)

  // Epilogue: bias + bf16 + scatter.
  const int cn = lane & 15;
  const int cm0 = (lane >> 4) << 3;
#pragma unroll
  for (int ms = 0; ms < 4; ++ms)
#pragma unroll
    for (int ns = 0; ns < 2; ++ns)
#pragma unroll
      for (int i = 0; i < 8; ++i) {
        int grow = m0 + wm * 64 + ms * 16 + cm0 + i;   // 0..4095
        int gcol = n0 + wn * 32 + ns * 16 + cn;        // 0..1023
        float v = acc[ms][ns][i] + bias[gcol];
        int bi = grow >> 10, rf = grow & 1023;
        int hi = rf >> 6,    r  = rf & 63;
        int c  = gcol >> 6,  di = gcol & 63;
        int si = r * 16 + c;
        size_t bh = (size_t)(bi * N_HEAD + hi);
        if (z == 2) {
          Vt[(bh * DHEAD + di) * SEQ + si] = (bf16)v;        // Vt[bh][d][s]
        } else {
          bf16* dst = (z == 0) ? Qh : Kh;
          dst[(bh * SEQ + si) * DHEAD + di] = (bf16)v;
        }
      }
}

// ---------------------------------------------------------------------------
// K2: per-row (b,h,q) max / sum-exp of causally-masked str_mat.
// ---------------------------------------------------------------------------
__global__ __launch_bounds__(256) void sm_stats_kernel(const float* __restrict__ strm,
                                                       float* __restrict__ smM,
                                                       float* __restrict__ smL)
{
  __shared__ float red[256];
  const size_t row = blockIdx.x;                 // 0..65535
  const int qi = (int)(row & (SEQ - 1));
  const float* p = strm + row * SEQ;
  const int tid = threadIdx.x;

  float m = -3.0e38f;
  for (int k = tid; k <= qi; k += 256) m = fmaxf(m, p[k]);
  red[tid] = m; __syncthreads();
  for (int s = 128; s > 0; s >>= 1) {
    if (tid < s) red[tid] = fmaxf(red[tid], red[tid + s]);
    __syncthreads();
  }
  const float rm = red[0];
  __syncthreads();

  float ssum = 0.0f;
  for (int k = tid; k <= qi; k += 256) ssum += __expf(p[k] - rm);
  red[tid] = ssum; __syncthreads();
  for (int s = 128; s > 0; s >>= 1) {
    if (tid < s) red[tid] += red[tid + s];
    __syncthreads();
  }
  if (tid == 0) { smM[row] = rm; smL[row] = red[0]; }
}

// ---------------------------------------------------------------------------
// K3: fused attention. One WG per (b*h, 16-row q tile).
//   Phase 1: scores(16x1024) = Q*K^T/64 + exp(str - m)/l  (masked -> +0)
//   Phase 2: full-row softmax in LDS, probs -> bf16
//   Phase 3: O(16x64) = P @ Vt via WMMA, key dim split over wave halves
//   Phase 4: un-reshape scatter into Ob[b,s,f] (bf16)
// Dynamic LDS: 64KB scores + 32KB probs + 8KB partials = 104KB (<320KB/WGP)
// ---------------------------------------------------------------------------
__global__ __launch_bounds__(256) void attn_kernel(
    const bf16* __restrict__ Qh, const bf16* __restrict__ Kh, const bf16* __restrict__ Vt,
    const float* __restrict__ strm,
    const float* __restrict__ smM, const float* __restrict__ smL,
    bf16* __restrict__ Ob)
{
  extern __shared__ char smem[];
  float* sc   = (float*)smem;                       // 16*1024 f32  (65536 B)
  bf16*  pr   = (bf16*)(smem + 65536);              // 16*1024 bf16 (32768 B)
  float* part = (float*)(smem + 65536 + 32768);     // 2*16*64 f32  (8192 B)
  __shared__ float red[256];
  __shared__ float rmax[16], rsum[16];

  const int bh = blockIdx.x;                        // 0..63
  const int q0 = blockIdx.y * 16;
  const int tid = threadIdx.x, wave = tid >> 5, lane = tid & 31;
  const bf16* Qg = Qh + (size_t)bh * SEQ * DHEAD;
  const bf16* Kg = Kh + (size_t)bh * SEQ * DHEAD;
  const bf16* Vg = Vt + (size_t)bh * DHEAD * SEQ;   // [d][s]

  // ---- Phase 1: S = Q K^T ; wave handles keys [wave*128, wave*128+128) ----
  v8f acc[8];
#pragma unroll
  for (int ns = 0; ns < 8; ++ns)
#pragma unroll
    for (int e = 0; e < 8; ++e) acc[ns][e] = 0.0f;

#pragma unroll
  for (int kk = 0; kk < DHEAD; kk += 32) {
    v16bf a = load_a16x32(Qg + (size_t)q0 * DHEAD + kk, DHEAD);
#pragma unroll
    for (int ns = 0; ns < 8; ++ns) {
      int key0 = wave * 128 + ns * 16;
      // element (k,n) = K[key0+n][kk+k]: K rows are contiguous d-vectors
      v16bf b = load_b32x16_T(Kg + (size_t)key0 * DHEAD + kk, DHEAD);
      acc[ns] = wmma_bf16(a, b, acc[ns]);
    }
  }

  {
    const int cn = lane & 15;
    const int cm0 = (lane >> 4) << 3;
#pragma unroll
    for (int i = 0; i < 8; ++i) {
      const int mrow = cm0 + i;
      const int qpos = q0 + mrow;
      const size_t rowid = (size_t)bh * SEQ + qpos;
      const float mM = smM[rowid];
      const float invL = 1.0f / smL[rowid];
      const float* srow = strm + rowid * SEQ;
#pragma unroll
      for (int ns = 0; ns < 8; ++ns) {
        const int key = wave * 128 + ns * 16 + cn;
        float smv = 0.0f;
        if (key <= qpos) smv = __expf(srow[key] - mM) * invL;   // second str pass
        sc[mrow * SEQ + key] = acc[ns][i] * (1.0f / 64.0f) + smv;
      }
    }
  }
  __syncthreads();

  // ---- Phase 2: row softmax over 16x1024 scores; probs to bf16 ----
  {
    const int row = tid >> 4;           // 0..15
    const int seg = tid & 15;           // column interleave: col = seg + 16*j
    float* scr = sc + row * SEQ;

    float mloc = -3.0e38f;
    for (int j = 0; j < 64; ++j) mloc = fmaxf(mloc, scr[seg + j * 16]);
    red[tid] = mloc; __syncthreads();
    if (tid < 16) {
      float mm = -3.0e38f;
      for (int s = 0; s < 16; ++s) mm = fmaxf(mm, red[tid * 16 + s]);
      rmax[tid] = mm;
    }
    __syncthreads();
    const float rm = rmax[row];

    float sloc = 0.0f;
    for (int j = 0; j < 64; ++j) sloc += __expf(scr[seg + j * 16] - rm);
    red[tid] = sloc; __syncthreads();
    if (tid < 16) {
      float ss = 0.0f;
      for (int s = 0; s < 16; ++s) ss += red[tid * 16 + s];
      rsum[tid] = ss;
    }
    __syncthreads();
    const float inv = 1.0f / rsum[row];
    for (int j = 0; j < 64; ++j) {
      int col = seg + j * 16;
      pr[row * SEQ + col] = (bf16)(__expf(scr[col] - rm) * inv);
    }
  }
  __syncthreads();

  // ---- Phase 3: O = P @ V ; wave -> (out cols ns*16, key half kh*512) ----
  {
    const int ns = wave & 3;
    const int kh = wave >> 2;
    v8f oacc;
#pragma unroll
    for (int e = 0; e < 8; ++e) oacc[e] = 0.0f;

    for (int k0 = kh * 512; k0 < kh * 512 + 512; k0 += 32) {
      v16bf a = load_a16x32(pr + k0, SEQ);
      // element (k,n) = V[key k][d n] = Vt[n][k]: contiguous along k
      v16bf b = load_b32x16_T(Vg + (size_t)(ns * 16) * SEQ + k0, SEQ);
      oacc = wmma_bf16(a, b, oacc);
    }
    const int cn = lane & 15;
    const int cm0 = (lane >> 4) << 3;
#pragma unroll
    for (int i = 0; i < 8; ++i)
      part[kh * 1024 + (cm0 + i) * 64 + ns * 16 + cn] = oacc[i];
  }
  __syncthreads();

  // ---- Phase 4: reduce halves, un-reshape, store bf16 ----
  {
    const int bi = bh >> 4, hi = bh & 15;
    for (int e = tid; e < 1024; e += 256) {
      const int mrow = e >> 6, di = e & 63;
      const float o = part[e] + part[1024 + e];
      const int si = q0 + mrow;
      const int rowf = hi * 64 + (si >> 4);
      const int colf = (si & 15) * 64 + di;
      Ob[((size_t)(bi * SEQ + rowf)) * FEAT + colf] = (bf16)o;
    }
  }
}

// ---------------------------------------------------------------------------
// K4: output projection. out = Ob @ Wo + bo (f32 out). Wo pre-transposed.
// ---------------------------------------------------------------------------
__global__ __launch_bounds__(256) void out_gemm_kernel(
    const bf16* __restrict__ Ab, const bf16* __restrict__ WoT,
    const float* __restrict__ bias, float* __restrict__ out)
{
  __shared__ bf16 Xs [2 * TILE_ELT];
  __shared__ bf16 WsT[2 * TILE_ELT];

  const int tid = threadIdx.x;
  const int wave = tid >> 5, lane = tid & 31;
  const int wm = wave >> 2, wn = wave & 3;
  const int m0 = blockIdx.x * 128;
  const int n0 = blockIdx.y * 128;

  v8f acc[4][2];
#pragma unroll
  for (int ms = 0; ms < 4; ++ms)
#pragma unroll
    for (int ns = 0; ns < 2; ++ns)
#pragma unroll
      for (int e = 0; e < 8; ++e) acc[ms][ns][e] = 0.0f;

  const bf16* Xg = Ab + (size_t)m0 * FEAT;
  const bf16* Wg = WoT + (size_t)n0 * FEAT;
  GEMM_MAIN(Xg, Wg, acc)

  const int cn = lane & 15;
  const int cm0 = (lane >> 4) << 3;
#pragma unroll
  for (int ms = 0; ms < 4; ++ms)
#pragma unroll
    for (int ns = 0; ns < 2; ++ns)
#pragma unroll
      for (int i = 0; i < 8; ++i) {
        int grow = m0 + wm * 64 + ms * 16 + cm0 + i;
        int gcol = n0 + wn * 32 + ns * 16 + cn;
        out[(size_t)grow * FEAT + gcol] = acc[ms][ns][i] + bias[gcol];
      }
}

// ---------------------------------------------------------------------------
// Host-side launcher. Workspace layout (needs >= ~49 MB):
//   [ 0,  8M) Xb bf16     [ 8,10M) WqT bf16  [10,12M) WkT bf16
//   [12,14M) WvT bf16     [14,16M) WoT bf16
//   [16,24M) Q head bf16  [24,32M) K head    [32,40M) V head TRANSPOSED [bh][d][s]
//   [40,48M) attn-out bf16 in [b,s,f] layout
//   [48M, +256K) sm row-max f32 ; [+256K, +512K) sm row-sumexp f32
// ---------------------------------------------------------------------------
extern "C" void kernel_launch(void* const* d_in, const int* in_sizes, int n_in,
                              void* d_out, int out_size, void* d_ws, size_t ws_size,
                              hipStream_t stream) {
  const float* x    = (const float*)d_in[0];
  const float* strm = (const float*)d_in[1];
  // d_in[2] attn_mask: causal tril, reproduced analytically (key <= query)
  const float* Wq = (const float*)d_in[3];  const float* bq = (const float*)d_in[4];
  const float* Wk = (const float*)d_in[5];  const float* bk = (const float*)d_in[6];
  const float* Wv = (const float*)d_in[7];  const float* bv = (const float*)d_in[8];
  const float* Wo = (const float*)d_in[9];  const float* bo = (const float*)d_in[10];
  float* out = (float*)d_out;

  char* ws = (char*)d_ws;
  const size_t MB = 1ull << 20;
  bf16* Xb  = (bf16*)(ws + 0 * MB);
  bf16* WqT = (bf16*)(ws + 8 * MB);
  bf16* WkT = (bf16*)(ws + 10 * MB);
  bf16* WvT = (bf16*)(ws + 12 * MB);
  bf16* WoT = (bf16*)(ws + 14 * MB);
  bf16* Qh  = (bf16*)(ws + 16 * MB);
  bf16* Kh  = (bf16*)(ws + 24 * MB);
  bf16* Vt  = (bf16*)(ws + 32 * MB);
  bf16* Ob  = (bf16*)(ws + 40 * MB);
  float* smM = (float*)(ws + 48 * MB);
  float* smL = (float*)(ws + 48 * MB + (1 << 18));

  const int nX = ROWS_TOTAL * FEAT;   // 4 Mi
  cvt_f32_bf16<<<1024, 256, 0, stream>>>(x, Xb, nX);
  cvt_transpose_f32_bf16<<<512, 256, 0, stream>>>(Wq, WqT);
  cvt_transpose_f32_bf16<<<512, 256, 0, stream>>>(Wk, WkT);
  cvt_transpose_f32_bf16<<<512, 256, 0, stream>>>(Wv, WvT);
  cvt_transpose_f32_bf16<<<512, 256, 0, stream>>>(Wo, WoT);

  qkv_gemm_kernel<<<dim3(ROWS_TOTAL / 128, FEAT / 128, 3), 256, 0, stream>>>(
      Xb, WqT, WkT, WvT, bq, bk, bv, Qh, Kh, Vt);

  sm_stats_kernel<<<BATCH * N_HEAD * SEQ, 256, 0, stream>>>(strm, smM, smL);

  const size_t attnShm = 65536 + 32768 + 8192;   // 104 KB dynamic LDS
  attn_kernel<<<dim3(BATCH * N_HEAD, SEQ / 16), 256, attnShm, stream>>>(
      Qh, Kh, Vt, strm, smM, smL, Ob);

  out_gemm_kernel<<<dim3(ROWS_TOTAL / 128, FEAT / 128), 256, 0, stream>>>(
      Ob, WoT, bo, out);
}
